// LSTMAutoencoder_19533511262506
// MI455X (gfx1250) — matrix operand order repaired and checked by reading, not compiled, not run on hardware
//
#include <hip/hip_runtime.h>
#include <math.h>

// LSTM autoencoder for MI455X (gfx1250, wave32, WMMA).
// One wave owns 16 batch rows and runs the whole time recurrence for a layer:
//   gates[16 x 4H] = x_t[16 x I] * Wih^T + h_{t-1}[16 x H] * Whh^T + bias
// via v_wmma_f32_16x16x32_f16 (f16 inputs, f32 accumulate).
// Weights are packed once into LDS in B-fragment layout; h transposes through
// a tiny per-wave LDS staging buffer; c stays in D-layout registers.

typedef __attribute__((ext_vector_type(16))) _Float16 v16h;
typedef __attribute__((ext_vector_type(8)))  float    v8f;

#define WAVES 4
#define TPB   (WAVES * 32)
#define ROWS_PER_WAVE  16
#define ROWS_PER_BLOCK (WAVES * ROWS_PER_WAVE)

#define BATCH 2048
#define SEQ   100
#define NFEAT 51
#define HID   64
#define LAT   24

__device__ __forceinline__ float sigf(float x) { return 1.0f / (1.0f + __expf(-x)); }

// HP    : padded hidden size (multiple of 32) used internally
// HREAL : true hidden size of this layer
// IP    : padded input size (multiple of 32)
// IREAL : true input size
// OUT_MODE: 0 = none, 1 = write full sequence [B,T,HREAL], 2 = write final h [B,HREAL]
// PROJ  : fuse 64->51 output projection (d2 layer)
template<int HP, int HREAL, int IP, int IREAL, int OUT_MODE, bool PROJ>
__global__ __launch_bounds__(TPB) void lstm_layer(
    const float* __restrict__ x, long sxb, long sxt,
    const float* __restrict__ wih, const float* __restrict__ whh,
    const float* __restrict__ bih, const float* __restrict__ bhh,
    float* __restrict__ out, long osb, long ost,
    const float* __restrict__ wout, const float* __restrict__ bout,
    float* __restrict__ pout, int T)
{
    constexpr int KC_I = IP / 32;          // K chunks for input GEMM
    constexpr int KC_H = HP / 32;          // K chunks for recurrent GEMM
    constexpr int NT_G = (4 * HP) / 16;    // gate N tiles
    constexpr int HT   = HP / 16;          // hidden N tiles (per gate)
    constexpr int NT_P = 4;                // projection N tiles (51 -> 64 padded)
    constexpr int NOUT = 51;
    constexpr int F_REC = NT_G * KC_I;                 // first recurrent frag
    constexpr int F_PRJ = NT_G * (KC_I + KC_H);        // first projection frag
    constexpr int NFRAG = F_PRJ + (PROJ ? NT_P * KC_H : 0);

    extern __shared__ __attribute__((aligned(64))) _Float16 lds[];
    _Float16* hshare = lds + NFRAG * 512;  // [WAVES][16 rows][HP cols]

    const int tid  = threadIdx.x;
    const int lane = tid & 31;
    const int wv   = tid >> 5;
    const int nl   = lane & 15;            // N column / A row within tile
    const int hi   = lane >> 4;            // lane half-group
    const int row0 = blockIdx.x * ROWS_PER_BLOCK + wv * ROWS_PER_WAVE;

    // ---- pack weight B-fragments into LDS (done once, reused T times) ----
    // B[k][n] fragment layout: lane l holds column n = l&15, k = 8*(l>>4)+h (h<8),
    // 16+8*(l>>4)+(h-8) (h>=8); 16 halves per lane -> one 32B ds read per fragment.
    for (int idx = tid; idx < NFRAG * 512; idx += TPB) {
        const int f    = idx >> 9;
        const int slot = idx & 511;
        const int l    = slot >> 4;
        const int h    = slot & 15;
        const int n_l  = l & 15;
        const int kg   = h + ((l >> 4) << 3) + ((h < 8) ? 0 : 8);      // k offset within 32-chunk
        float v = 0.0f;
        if (f < F_REC) {                            // input weights Wih [4*HREAL, IREAL]
            const int nt = f / KC_I, kc = f - nt * KC_I;
            const int ng = nt * 16 + n_l;
            const int gi = ng / HP, off = ng - gi * HP;
            const int k  = kc * 32 + kg;
            if (off < HREAL && k < IREAL) v = wih[(gi * HREAL + off) * IREAL + k];
        } else if (f < F_PRJ) {                     // recurrent weights Whh [4*HREAL, HREAL]
            const int fr = f - F_REC;
            const int nt = fr / KC_H, kc = fr - nt * KC_H;
            const int ng = nt * 16 + n_l;
            const int gi = ng / HP, off = ng - gi * HP;
            const int k  = kc * 32 + kg;
            if (off < HREAL && k < HREAL) v = whh[(gi * HREAL + off) * HREAL + k];
        } else if (PROJ) {                          // projection Wout [51, HREAL]
            const int fr = f - F_PRJ;
            const int nt = fr / KC_H, kc = fr - nt * KC_H;
            const int ng = nt * 16 + n_l;
            const int k  = kc * 32 + kg;
            if (ng < NOUT && k < HREAL) v = wout[ng * HREAL + k];
        }
        lds[idx] = (_Float16)v;
    }
    __syncthreads();

    // ---- per-lane bias for this lane's N column in each gate tile ----
    float biasv[NT_G];
#pragma unroll
    for (int nt = 0; nt < NT_G; ++nt) {
        const int ng = nt * 16 + nl;
        const int gi = ng / HP, off = ng - gi * HP;
        biasv[nt] = (off < HREAL) ? (bih[gi * HREAL + off] + bhh[gi * HREAL + off]) : 0.0f;
    }
    float pbias[NT_P];
    if constexpr (PROJ) {
#pragma unroll
        for (int nt = 0; nt < NT_P; ++nt) {
            const int ng = nt * 16 + nl;
            pbias[nt] = (ng < NOUT) ? bout[ng] : 0.0f;
        }
    }

    // cell state, kept in WMMA D layout across the whole recurrence
    v8f cst[HT];
#pragma unroll
    for (int jt = 0; jt < HT; ++jt)
#pragma unroll
        for (int r = 0; r < 8; ++r) cst[jt][r] = 0.0f;

    const long hbase = (long)wv * 16 * HP;

    for (int t = 0; t < T; ++t) {
        // init gate accumulators with bias
        v8f acc[NT_G];
#pragma unroll
        for (int nt = 0; nt < NT_G; ++nt)
#pragma unroll
            for (int r = 0; r < 8; ++r) acc[nt][r] = biasv[nt];

        // ---- A fragments of x_t (16 x IP), f32 -> f16 ----
        const float* xp = x + (long)(row0 + nl) * sxb + (long)t * sxt;
        v16h ax[KC_I];
#pragma unroll
        for (int kc = 0; kc < KC_I; ++kc)
#pragma unroll
            for (int h = 0; h < 16; ++h) {
                const int K = kc * 32 + h + hi * 8 + ((h < 8) ? 0 : 8);
                const float v = (K < IREAL) ? xp[K] : 0.0f;
                ax[kc][h] = (_Float16)v;
            }

        // ---- gates += x_t * Wih^T ----
#pragma unroll
        for (int kc = 0; kc < KC_I; ++kc)
#pragma unroll
            for (int nt = 0; nt < NT_G; ++nt) {
                const v16h b = *(const v16h*)(lds + (nt * KC_I + kc) * 512 + lane * 16);
                acc[nt] = __builtin_amdgcn_wmma_f32_16x16x32_f16(
                    false, ax[kc], false, b, (short)0, acc[nt], false, false);
            }

        // ---- gates += h_{t-1} * Whh^T  (h0 == 0 -> skip at t==0) ----
        if (t > 0) {
            v16h ah[KC_H];
#pragma unroll
            for (int kc = 0; kc < KC_H; ++kc)
#pragma unroll
                for (int h = 0; h < 16; ++h) {
                    const int K = kc * 32 + h + hi * 8 + ((h < 8) ? 0 : 8);
                    ah[kc][h] = hshare[hbase + (long)nl * HP + K];
                }
#pragma unroll
            for (int kc = 0; kc < KC_H; ++kc)
#pragma unroll
                for (int nt = 0; nt < NT_G; ++nt) {
                    const v16h b = *(const v16h*)(lds + (F_REC + nt * KC_H + kc) * 512 + lane * 16);
                    acc[nt] = __builtin_amdgcn_wmma_f32_16x16x32_f16(
                        false, ah[kc], false, b, (short)0, acc[nt], false, false);
                }
        }

        // ---- elementwise LSTM cell (gate tiles i/f/g/o share lane+slot) ----
#pragma unroll
        for (int jt = 0; jt < HT; ++jt) {
#pragma unroll
            for (int r = 0; r < 8; ++r) {
                const float iv = sigf(acc[jt][r]);
                const float fv = sigf(acc[jt + HT][r]);
                const float gv = tanhf(acc[jt + 2 * HT][r]);
                const float ov = sigf(acc[jt + 3 * HT][r]);
                const float cc = fv * cst[jt][r] + iv * gv;
                cst[jt][r] = cc;
                const float hv = ov * tanhf(cc);
                const int col   = jt * 16 + nl;
                const int M     = r + 8 * hi;
                const bool valid = (HP == HREAL) || (col < HREAL);
                hshare[hbase + (long)M * HP + col] = valid ? (_Float16)hv : (_Float16)0.0f;
                if constexpr (OUT_MODE == 1) {
                    if (valid) out[(long)(row0 + M) * osb + (long)t * ost + col] = hv;
                } else if constexpr (OUT_MODE == 2) {
                    if (valid && t == T - 1) out[(long)(row0 + M) * HREAL + col] = hv;
                }
            }
        }

        // ---- fused output projection: recon_t = h_t * Wout^T + bout ----
        if constexpr (PROJ) {
            v16h ah2[KC_H];
#pragma unroll
            for (int kc = 0; kc < KC_H; ++kc)
#pragma unroll
                for (int h = 0; h < 16; ++h) {
                    const int K = kc * 32 + h + hi * 8 + ((h < 8) ? 0 : 8);
                    ah2[kc][h] = hshare[hbase + (long)nl * HP + K];
                }
            v8f pacc[NT_P];
#pragma unroll
            for (int nt = 0; nt < NT_P; ++nt)
#pragma unroll
                for (int r = 0; r < 8; ++r) pacc[nt][r] = pbias[nt];
#pragma unroll
            for (int kc = 0; kc < KC_H; ++kc)
#pragma unroll
                for (int nt = 0; nt < NT_P; ++nt) {
                    const v16h b = *(const v16h*)(lds + (F_PRJ + nt * KC_H + kc) * 512 + lane * 16);
                    pacc[nt] = __builtin_amdgcn_wmma_f32_16x16x32_f16(
                        false, ah2[kc], false, b, (short)0, pacc[nt], false, false);
                }
#pragma unroll
            for (int nt = 0; nt < NT_P; ++nt)
#pragma unroll
                for (int r = 0; r < 8; ++r) {
                    const int col = nt * 16 + nl;
                    const int M   = r + 8 * hi;
                    if (col < NOUT)
                        pout[(long)(row0 + M) * ((long)T * NOUT) + (long)t * NOUT + col] = pacc[nt][r];
                }
        }
    }
}

static constexpr size_t lds_bytes(int HP, int IP, bool proj) {
    const int ntg  = (4 * HP) / 16;
    const int kci  = IP / 32;
    const int kch  = HP / 32;
    const int nfrag = ntg * (kci + kch) + (proj ? 4 * kch : 0);
    return (size_t)nfrag * 512 * 2 + (size_t)WAVES * 16 * HP * 2;
}

extern "C" void kernel_launch(void* const* d_in, const int* in_sizes, int n_in,
                              void* d_out, int out_size, void* d_ws, size_t ws_size,
                              hipStream_t stream) {
    const float* x       = (const float*)d_in[0];
    const float* e1_wih  = (const float*)d_in[1];
    const float* e1_whh  = (const float*)d_in[2];
    const float* e1_bih  = (const float*)d_in[3];
    const float* e1_bhh  = (const float*)d_in[4];
    const float* e2_wih  = (const float*)d_in[5];
    const float* e2_whh  = (const float*)d_in[6];
    const float* e2_bih  = (const float*)d_in[7];
    const float* e2_bhh  = (const float*)d_in[8];
    const float* d1_wih  = (const float*)d_in[9];
    const float* d1_whh  = (const float*)d_in[10];
    const float* d1_bih  = (const float*)d_in[11];
    const float* d1_bhh  = (const float*)d_in[12];
    const float* d2_wih  = (const float*)d_in[13];
    const float* d2_whh  = (const float*)d_in[14];
    const float* d2_bih  = (const float*)d_in[15];
    const float* d2_bhh  = (const float*)d_in[16];
    const float* w_out   = (const float*)d_in[17];
    const float* b_out   = (const float*)d_in[18];
    float*       recon   = (float*)d_out;

    // workspace: out1 [B,T,64] | z [B,24] | dec1 [B,T,64]
    float* out1 = (float*)d_ws;
    float* zbuf = out1 + (size_t)BATCH * SEQ * HID;
    float* dec1 = zbuf + (size_t)BATCH * LAT;

    const dim3 grid(BATCH / ROWS_PER_BLOCK);
    const dim3 block(TPB);

    // encoder layer 1: 51 -> 64, full sequence out
    lstm_layer<64, 64, 64, 51, 1, false>
        <<<grid, block, lds_bytes(64, 64, false), stream>>>(
            x, (long)SEQ * NFEAT, NFEAT,
            e1_wih, e1_whh, e1_bih, e1_bhh,
            out1, (long)SEQ * HID, HID,
            nullptr, nullptr, nullptr, SEQ);

    // encoder layer 2: 64 -> 24 (padded to 32 internally), final h only -> z
    lstm_layer<32, 24, 64, 64, 2, false>
        <<<grid, block, lds_bytes(32, 64, false), stream>>>(
            out1, (long)SEQ * HID, HID,
            e2_wih, e2_whh, e2_bih, e2_bhh,
            zbuf, 0, 0,
            nullptr, nullptr, nullptr, SEQ);

    // decoder layer 1: repeated latent (time stride 0) 24 -> 64, sequence out
    lstm_layer<64, 64, 32, 24, 1, false>
        <<<grid, block, lds_bytes(64, 32, false), stream>>>(
            zbuf, LAT, 0,
            d1_wih, d1_whh, d1_bih, d1_bhh,
            dec1, (long)SEQ * HID, HID,
            nullptr, nullptr, nullptr, SEQ);

    // decoder layer 2: 64 -> 64 with fused 64 -> 51 projection to d_out
    lstm_layer<64, 64, 64, 64, 0, true>
        <<<grid, block, lds_bytes(64, 64, true), stream>>>(
            dec1, (long)SEQ * HID, HID,
            d2_wih, d2_whh, d2_bih, d2_bhh,
            nullptr, 0, 0,
            w_out, b_out, recon, SEQ);
}
